// TemporalConvAttention_57690000720289
// MI455X (gfx1250) — matrix operand order repaired and checked by reading, not compiled, not run on hardware
//
#include <hip/hip_runtime.h>
#include <math.h>

typedef _Float16 f16;
typedef __attribute__((ext_vector_type(16))) _Float16 v16h;
typedef __attribute__((ext_vector_type(8)))  _Float16 h8;
typedef __attribute__((ext_vector_type(8)))  float    v8f;

#define EMBED 1024
#define NHEAD 16
#define HDIM  64
#define BB    2
#define SS    2048
#define TOK   (BB*SS)   // 4096

// ---------------- WMMA fragment helpers (16x16x32 f16, this file's contiguous k = 16h + s order) ----------------
// A-matrix 16x32: lanes 0-15 hold row M=lane, K=k0..k0+15;
//                 lanes 16-31 hold row M=lane-16, K=k0+16..k0+31.
__device__ __forceinline__ v16h load_a_frag(const f16* A, int lda, int mbase, int k0, int lane) {
  int row = mbase + (lane & 15);
  int kk  = k0 + ((lane >> 4) << 4);
  const f16* p = A + (size_t)row * lda + kk;
  h8 lo = *(const h8*)p;
  h8 hi = *(const h8*)(p + 8);
  v16h r;
#pragma unroll
  for (int i = 0; i < 8; ++i) { r[i] = lo[i]; r[8 + i] = hi[i]; }
  return r;
}

// B-matrix 32x16 (K x N): lanes 0-15 hold col N=lane, K=k0..k0+15;
//                         lanes 16-31 hold col N=lane-16, K=k0+16..k0+31.
// W is row-major [N, Ktot] (torch [out,in]) -> contiguous 32B load per lane.
__device__ __forceinline__ v16h load_b_frag(const f16* W, int ldw, int nbase, int k0, int lane) {
  int col = nbase + (lane & 15);
  int kk  = k0 + ((lane >> 4) << 4);
  return *(const v16h*)(W + (size_t)col * ldw + kk);
}

__device__ __forceinline__ v8f wmma_f16(v16h a, v16h b, v8f c) {
  return __builtin_amdgcn_wmma_f32_16x16x32_f16(false, a, false, b, (short)0, c, false, false);
}

// ---------------- small prep kernels ----------------
__global__ void f32_to_f16_kernel(const float* __restrict__ in, f16* __restrict__ out, int n) {
  int i = blockIdx.x * blockDim.x + threadIdx.x;
  if (i < n) out[i] = (f16)in[i];
}

// depthwise conv over time (K=3, zero pad), bias, emit f16 [B*S, d]
__global__ void dwconv_kernel(const float* __restrict__ x, const float* __restrict__ w,
                              const float* __restrict__ bias, f16* __restrict__ out) {
  int i = blockIdx.x * blockDim.x + threadIdx.x;   // over TOK*EMBED
  int c = i & (EMBED - 1);
  int s = (i >> 10) & (SS - 1);
  const float* xp = x + (size_t)i;
  float v = xp[0] * w[c * 3 + 1] + bias[c];
  if (s > 0)      v += xp[-EMBED] * w[c * 3 + 0];
  if (s < SS - 1) v += xp[ EMBED] * w[c * 3 + 2];
  out[i] = (f16)v;
}

// V [B*S, d] f16 -> Vt [(b*H+h)][hd][S] f16 so PV B-fragments are contiguous
__global__ void vtrans_kernel(const f16* __restrict__ V, f16* __restrict__ Vt) {
  int i = blockIdx.x * blockDim.x + threadIdx.x;   // B*H*HDIM*SS = 4194304
  int s = i & (SS - 1);
  int n = (i >> 11) & (HDIM - 1);
  int h = (i >> 17) & (NHEAD - 1);
  int b = i >> 21;
  Vt[i] = V[((size_t)b * SS + s) * EMBED + h * HDIM + n];
}

// ---------------- tiled WMMA GEMM: out[M,N] = A[M,K] * W[N,K]^T + bias ----------------
// block = 8 waves, wave tile 32x64 (2x4 of 16x16), block tile 128x128
template<bool GELU, bool HALF_OUT>
__global__ __launch_bounds__(256) void gemm_kernel(
    const f16* __restrict__ A, const f16* __restrict__ W, const float* __restrict__ bias,
    f16* __restrict__ outh, float* __restrict__ outf, int ldo, int Kdim) {
  int lane  = threadIdx.x & 31;
  int wave  = threadIdx.x >> 5;
  int mbase = blockIdx.x * 128 + (wave & 3) * 32;
  int nbase = blockIdx.y * 128 + (wave >> 2) * 64;

  v8f acc[2][4] = {};
  for (int k0 = 0; k0 < Kdim; k0 += 32) {
    if (k0 + 32 < Kdim) {  // hint next K-slab into cache (global_prefetch_b8)
      __builtin_prefetch(A + (size_t)(mbase + (lane & 15)) * Kdim + k0 + 32, 0, 1);
      __builtin_prefetch(W + (size_t)(nbase + (lane & 15)) * Kdim + k0 + 32, 0, 1);
    }
    v16h a0 = load_a_frag(A, Kdim, mbase,      k0, lane);
    v16h a1 = load_a_frag(A, Kdim, mbase + 16, k0, lane);
    v16h b0 = load_b_frag(W, Kdim, nbase,      k0, lane);
    v16h b1 = load_b_frag(W, Kdim, nbase + 16, k0, lane);
    v16h b2 = load_b_frag(W, Kdim, nbase + 32, k0, lane);
    v16h b3 = load_b_frag(W, Kdim, nbase + 48, k0, lane);
    acc[0][0] = wmma_f16(a0, b0, acc[0][0]);
    acc[0][1] = wmma_f16(a0, b1, acc[0][1]);
    acc[0][2] = wmma_f16(a0, b2, acc[0][2]);
    acc[0][3] = wmma_f16(a0, b3, acc[0][3]);
    acc[1][0] = wmma_f16(a1, b0, acc[1][0]);
    acc[1][1] = wmma_f16(a1, b1, acc[1][1]);
    acc[1][2] = wmma_f16(a1, b2, acc[1][2]);
    acc[1][3] = wmma_f16(a1, b3, acc[1][3]);
  }

  int rq = (lane >> 4) << 3;   // C/D layout: VGPR j -> (M=j+rq, N=lane&15)
#pragma unroll
  for (int mi = 0; mi < 2; ++mi) {
#pragma unroll
    for (int t = 0; t < 4; ++t) {
      int col = nbase + 16 * t + (lane & 15);
      float bv = bias[col];
#pragma unroll
      for (int j = 0; j < 8; ++j) {
        int row = mbase + 16 * mi + rq + j;
        float v = acc[mi][t][j] + bv;
        if (GELU) v = 0.5f * v * (1.0f + erff(v * 0.70710678118654752f));
        if (HALF_OUT) outh[(size_t)row * ldo + col] = (f16)v;
        else          outf[(size_t)row * ldo + col] = v;
      }
    }
  }
}

// ---------------- flash attention: one wave per 16-query tile ----------------
__global__ __launch_bounds__(256) void attn_kernel(
    const f16* __restrict__ Q, const f16* __restrict__ Kmat,
    const f16* __restrict__ Vt, f16* __restrict__ O) {
  __shared__ f16 psh[8][16 * 32];     // per-wave P staging for the layout transpose
  int lane = threadIdx.x & 31;
  int wave = threadIdx.x >> 5;
  int tile = blockIdx.x * 8 + wave;   // B*H*(S/16) = 4096 tiles
  int qt = tile & 127;                // S/16 = 128
  int bh = tile >> 7;                 // b*16 + h
  int h  = bh & (NHEAD - 1);
  int b  = bh >> 4;

  const f16* Qp = Q    + (size_t)b * SS * EMBED + h * HDIM;
  const f16* Kp = Kmat + (size_t)b * SS * EMBED + h * HDIM;
  const f16* Vp = Vt   + (size_t)bh * HDIM * SS;
  int qbase = qt * 16;
  int rq = (lane >> 4) << 3;

  v16h qf0 = load_a_frag(Qp, EMBED, qbase, 0,  lane);
  v16h qf1 = load_a_frag(Qp, EMBED, qbase, 32, lane);

  v8f acc[4] = {};
  float m_i[8], l_i[8];
#pragma unroll
  for (int j = 0; j < 8; ++j) { m_i[j] = -1e30f; l_i[j] = 0.f; }
  f16* sp = &psh[wave][0];

  for (int c = 0; c < SS; c += 32) {
    // scores for 32 keys = two 16x16 tiles, K-dim 64 = two WMMA steps each
    v8f s0 = {}, s1 = {};
    s0 = wmma_f16(qf0, load_b_frag(Kp, EMBED, c,      0,  lane), s0);
    s0 = wmma_f16(qf1, load_b_frag(Kp, EMBED, c,      32, lane), s0);
    s1 = wmma_f16(qf0, load_b_frag(Kp, EMBED, c + 16, 0,  lane), s1);
    s1 = wmma_f16(qf1, load_b_frag(Kp, EMBED, c + 16, 32, lane), s1);

    float p0[8], p1[8];
#pragma unroll
    for (int j = 0; j < 8; ++j) {
      float a  = s0[j] * 0.125f;      // 1/sqrt(64)
      float bb = s1[j] * 0.125f;
      float mx = fmaxf(a, bb);
#pragma unroll
      for (int off = 8; off >= 1; off >>= 1) mx = fmaxf(mx, __shfl_xor(mx, off, 16));
      float mnew  = fmaxf(m_i[j], mx);
      float ratio = __expf(m_i[j] - mnew);
      float e0 = __expf(a - mnew), e1 = __expf(bb - mnew);
      float rs = e0 + e1;
#pragma unroll
      for (int off = 8; off >= 1; off >>= 1) rs += __shfl_xor(rs, off, 16);
      l_i[j] = l_i[j] * ratio + rs;
      m_i[j] = mnew;
      p0[j] = e0; p1[j] = e1;
#pragma unroll
      for (int t = 0; t < 4; ++t) acc[t][j] *= ratio;
    }

    // C-layout -> A-layout transpose of P through LDS (DS ops are wave-in-order)
#pragma unroll
    for (int j = 0; j < 8; ++j) {
      sp[(rq + j) * 32 + (lane & 15)]      = (f16)p0[j];
      sp[(rq + j) * 32 + (lane & 15) + 16] = (f16)p1[j];
    }
    asm volatile("s_wait_dscnt 0x0" ::: "memory");
    v16h pf = load_a_frag(sp, 32, 0, 0, lane);

#pragma unroll
    for (int t = 0; t < 4; ++t)
      acc[t] = wmma_f16(pf, load_b_frag(Vp, SS, 16 * t, c, lane), acc[t]);
  }

#pragma unroll
  for (int t = 0; t < 4; ++t) {
    int col = 16 * t + (lane & 15);
#pragma unroll
    for (int j = 0; j < 8; ++j) {
      int row = qbase + rq + j;
      O[((size_t)b * SS + row) * EMBED + h * HDIM + col] = (f16)(acc[t][j] / l_i[j]);
    }
  }
}

// ---------------- launcher ----------------
extern "C" void kernel_launch(void* const* d_in, const int* in_sizes, int n_in,
                              void* d_out, int out_size, void* d_ws, size_t ws_size,
                              hipStream_t stream) {
  const float* x    = (const float*)d_in[0];
  const float* wq   = (const float*)d_in[1];
  const float* bq   = (const float*)d_in[2];
  const float* wk   = (const float*)d_in[3];
  const float* bk   = (const float*)d_in[4];
  const float* wv   = (const float*)d_in[5];
  const float* bv   = (const float*)d_in[6];
  const float* wo   = (const float*)d_in[7];
  const float* bo   = (const float*)d_in[8];
  const float* dw_w = (const float*)d_in[9];
  const float* dw_b = (const float*)d_in[10];
  const float* pw_w = (const float*)d_in[11];
  const float* pw_b = (const float*)d_in[12];
  const float* fu_w = (const float*)d_in[13];
  const float* fu_b = (const float*)d_in[14];
  float* out = (float*)d_out;

  char* ws = (char*)d_ws;
  size_t off = 0;
  auto alloc = [&](size_t elems) -> f16* {
    f16* p = (f16*)(ws + off);
    off = (off + elems * sizeof(f16) + 255) & ~((size_t)255);
    return p;
  };
  const size_t DD = (size_t)EMBED * EMBED;
  f16* xb     = alloc((size_t)TOK * EMBED);
  f16* wqh    = alloc(DD);
  f16* wkh    = alloc(DD);
  f16* wvh    = alloc(DD);
  f16* woh    = alloc(DD);
  f16* pwh    = alloc(DD);
  f16* fuh    = alloc(2 * DD);
  f16* cb     = alloc((size_t)TOK * EMBED);
  f16* Qh     = alloc((size_t)TOK * EMBED);
  f16* Kh     = alloc((size_t)TOK * EMBED);
  f16* Vh     = alloc((size_t)TOK * EMBED);
  f16* Vt     = alloc((size_t)TOK * EMBED);
  f16* fusedh = alloc((size_t)TOK * 2 * EMBED);
  f16* AOh    = Vh;   // V (row-major) is dead after vtrans; reuse for attention output

  const int nTokEl = TOK * EMBED;                 // 4194304
  const int nW  = EMBED * EMBED;
  f32_to_f16_kernel<<<(nTokEl + 255) / 256, 256, 0, stream>>>(x, xb, nTokEl);
  f32_to_f16_kernel<<<(nW + 255) / 256, 256, 0, stream>>>(wq, wqh, nW);
  f32_to_f16_kernel<<<(nW + 255) / 256, 256, 0, stream>>>(wk, wkh, nW);
  f32_to_f16_kernel<<<(nW + 255) / 256, 256, 0, stream>>>(wv, wvh, nW);
  f32_to_f16_kernel<<<(nW + 255) / 256, 256, 0, stream>>>(wo, woh, nW);
  f32_to_f16_kernel<<<(nW + 255) / 256, 256, 0, stream>>>(pw_w, pwh, nW);
  f32_to_f16_kernel<<<(2 * nW + 255) / 256, 256, 0, stream>>>(fu_w, fuh, 2 * nW);

  dwconv_kernel<<<nTokEl / 256, 256, 0, stream>>>(x, dw_w, dw_b, cb);

  dim3 g1(TOK / 128, EMBED / 128);                // 32 x 8
  gemm_kernel<false, true><<<g1, 256, 0, stream>>>(xb, wqh, bq, Qh, nullptr, EMBED, EMBED);
  gemm_kernel<false, true><<<g1, 256, 0, stream>>>(xb, wkh, bk, Kh, nullptr, EMBED, EMBED);
  gemm_kernel<false, true><<<g1, 256, 0, stream>>>(xb, wvh, bv, Vh, nullptr, EMBED, EMBED);
  vtrans_kernel<<<nTokEl / 256, 256, 0, stream>>>(Vh, Vt);

  // conv branch: pointwise GEMM + exact GELU -> right half of fusion buffer
  gemm_kernel<true, true><<<g1, 256, 0, stream>>>(cb, pwh, pw_b, fusedh + EMBED, nullptr, 2 * EMBED, EMBED);

  attn_kernel<<<(BB * NHEAD * (SS / 16)) / 8, 256, 0, stream>>>(Qh, Kh, Vt, AOh);

  // attention output projection -> left half of fusion buffer
  gemm_kernel<false, true><<<g1, 256, 0, stream>>>(AOh, woh, bo, fusedh, nullptr, 2 * EMBED, EMBED);

  // fusion GEMM over K = 2d, fp32 output
  gemm_kernel<false, false><<<g1, 256, 0, stream>>>(fusedh, fuh, fu_b, nullptr, out, EMBED, 2 * EMBED);

  (void)in_sizes; (void)n_in; (void)out_size; (void)ws_size;
}
